// GNAH_27041114096438
// MI455X (gfx1250) — matrix operand order repaired and checked by reading, not compiled, not run on hardware
//
#include <hip/hip_runtime.h>
#include <hip/hip_bf16.h>
#include <math.h>

typedef __attribute__((ext_vector_type(16))) __bf16 v16bf;
typedef __attribute__((ext_vector_type(8)))  __bf16 v8bf;
typedef __attribute__((ext_vector_type(8)))  float  v8f;
typedef __attribute__((ext_vector_type(4)))  float  v4f;

#define N_NODES 1024
#define E_DIM   128
#define NEG_SLOPE 0.2f
#define JT 128          // j-rows per block (8 waves * 16)
#define BK_STRIDE 136   // padded bf16 row stride for transposed B in LDS

__device__ __forceinline__ float lrelu(float x) { return x >= 0.f ? x : NEG_SLOPE * x; }

// ---------------------------------------------------------------------------
// Kernel A: per-row precompute.
//   wf[j]   = embs[j] @ W
//   cb[j,l] = (wf[j] @ pc_Wc)[l] + pc_b[l]
//   wfa1[j] = wf[j] . a_W[:E]
//   Wa2[k]  = sum_l W[k,l] * a_W[E+l]      (block 0 only)
// ---------------------------------------------------------------------------
__global__ __launch_bounds__(128) void setup_kernel(
    const float* __restrict__ embs, const float* __restrict__ W,
    const float* __restrict__ pc_W, const float* __restrict__ pc_b,
    const float* __restrict__ a_W,
    float* __restrict__ cb, float* __restrict__ wfa1, float* __restrict__ Wa2)
{
    __shared__ float s_emb[E_DIM];
    __shared__ float s_wf[E_DIM];
    __shared__ float s_red[E_DIM];

    const int j = blockIdx.x;
    const int l = threadIdx.x;

    s_emb[l] = embs[j * E_DIM + l];
    __syncthreads();

    float wf = 0.f;
    #pragma unroll 4
    for (int k = 0; k < E_DIM; ++k) wf = fmaf(s_emb[k], W[k * E_DIM + l], wf);
    s_wf[l] = wf;
    __syncthreads();

    float c = pc_b[l];
    #pragma unroll 4
    for (int k = 0; k < E_DIM; ++k) c = fmaf(s_wf[k], pc_W[k * E_DIM + l], c);
    cb[j * E_DIM + l] = c;

    s_red[l] = wf * a_W[l];
    __syncthreads();
    for (int s = 64; s > 0; s >>= 1) {
        if (l < s) s_red[l] += s_red[l + s];
        __syncthreads();
    }
    if (l == 0) wfa1[j] = s_red[0];

    if (blockIdx.x == 0) {
        float w2 = 0.f;
        #pragma unroll 4
        for (int t = 0; t < E_DIM; ++t) w2 = fmaf(W[l * E_DIM + t], a_W[E_DIM + t], w2);
        Wa2[l] = w2;
    }
}

// ---------------------------------------------------------------------------
// Kernel B (hot): logits[i,j] via bf16 WMMA over pde, fully fused.
// Grid: (N/JT, N) = (8, 1024); 256 threads = 8 waves; wave w owns j-rows
// j0 + 16w .. +15. pde slice (128x128 f32) read exactly once, converted to
// bf16 in-register for v_wmma_f32_16x16x32_bf16.
// ---------------------------------------------------------------------------
__global__ __launch_bounds__(256) void logits_kernel(
    const float* __restrict__ pde, const float* __restrict__ pc_Wp,
    const float* __restrict__ cb,  const float* __restrict__ wfa1,
    const float* __restrict__ Wa2, const float* __restrict__ a_b,
    float* __restrict__ logits)
{
    __shared__ __bf16 BsT[E_DIM * BK_STRIDE];  // pc_Wp transposed: [l][k]
    __shared__ float  s_Wa2[E_DIM];

    const int i   = blockIdx.y;
    const int j0  = blockIdx.x * JT;
    const int tid = threadIdx.x;

    // Stage pc_Wp^T into LDS as bf16 (coalesced global reads, L2-hot).
    {
        const int l = tid & 127;
        for (int k = tid >> 7; k < E_DIM; k += 2)
            BsT[l * BK_STRIDE + k] = (__bf16)pc_Wp[k * E_DIM + l];
    }
    if (tid < E_DIM) s_Wa2[tid] = Wa2[tid];
    __syncthreads();

    const int wave = tid >> 5;
    const int lane = tid & 31;
    const int hi   = lane >> 4;   // half-wave select
    const int l16  = lane & 15;
    const int jw   = j0 + wave * 16;

    // A fragments: 16x32 bf16 layout (ISA 7.12.2):
    //   lanes 0-15 : row M=l16, K = {0..7, 16..23} (+32s)
    //   lanes 16-31: row M=l16, K = {8..15, 24..31} (+32s)
    const float* prow = pde + ((size_t)i * N_NODES + (jw + l16)) * E_DIM;
    v16bf a[4];
    #pragma unroll
    for (int s = 0; s < 4; ++s) {
        const int kb1 = 32 * s + (hi ? 8 : 0);
        const int kb2 = kb1 + 16;
        v4f f0 = *(const v4f*)(prow + kb1);
        v4f f1 = *(const v4f*)(prow + kb1 + 4);
        v4f f2 = *(const v4f*)(prow + kb2);
        v4f f3 = *(const v4f*)(prow + kb2 + 4);
        v16bf t;
        #pragma unroll
        for (int e = 0; e < 4; ++e) {
            t[e]      = (__bf16)f0[e];
            t[4 + e]  = (__bf16)f1[e];
            t[8 + e]  = (__bf16)f2[e];
            t[12 + e] = (__bf16)f3[e];
        }
        a[s] = t;
    }

    float psum[8] = {0.f, 0.f, 0.f, 0.f, 0.f, 0.f, 0.f, 0.f};

    #pragma unroll
    for (int n = 0; n < 8; ++n) {   // 8 column tiles of l
        const int ncol = n * 16 + l16;
        // B fragment 32x16 bf16: col N=l16; lane<16 -> K={0..7,16..23}; lane>=16 -> K={8..15,24..31}.
        const __bf16* bp = &BsT[ncol * BK_STRIDE + hi * 8];
        v8f acc = {};
        #pragma unroll
        for (int s = 0; s < 4; ++s) {
            v8bf blo = *(const v8bf*)(bp + 32 * s);
            v8bf bhi = *(const v8bf*)(bp + 32 * s + 16);
            v16bf b;
            #pragma unroll
            for (int e = 0; e < 8; ++e) { b[e] = blo[e]; b[8 + e] = bhi[e]; }
            acc = __builtin_amdgcn_wmma_f32_16x16x32_bf16(
                false, a[s], false, b, (short)0, acc, false, false);
        }
        // acc[r] = (pde[i, jw + r + 8*hi, :] . pc_Wp[:, ncol])
        const float w2 = s_Wa2[ncol];
        #pragma unroll
        for (int r = 0; r < 8; ++r) {
            const int j = jw + r + 8 * hi;
            const float val = cb[j * E_DIM + ncol] + acc[r];
            psum[r] = fmaf(lrelu(val), w2, psum[r]);
        }
    }

    // Reduce each psum[r] across its 16-lane half-wave; lane l16==0 writes.
    const float bias = wfa1[i] + a_b[0];
    #pragma unroll
    for (int r = 0; r < 8; ++r) {
        float v = psum[r];
        v += __shfl_xor(v, 1, 32);
        v += __shfl_xor(v, 2, 32);
        v += __shfl_xor(v, 4, 32);
        v += __shfl_xor(v, 8, 32);
        if (l16 == 0) {
            const int j = jw + r + 8 * hi;
            logits[(size_t)i * N_NODES + j] = lrelu(bias + v);
        }
    }
}

// ---------------------------------------------------------------------------
// Kernel C: masked softmax over axis=1, in place (logits -> attn).
// ---------------------------------------------------------------------------
__global__ __launch_bounds__(256) void softmax_kernel(
    const int* __restrict__ adj, float* __restrict__ logits)
{
    __shared__ float red[256];
    const int i = blockIdx.x;
    const int t = threadIdx.x;
    const int*  arow = adj    + (size_t)i * N_NODES;
    float*      lrow = logits + (size_t)i * N_NODES;

    float m = -INFINITY;
    for (int j = t; j < N_NODES; j += 256) {
        float v = (arow[j] == 1) ? lrow[j] : -INFINITY;
        m = fmaxf(m, v);
    }
    red[t] = m; __syncthreads();
    for (int s = 128; s > 0; s >>= 1) {
        if (t < s) red[t] = fmaxf(red[t], red[t + s]);
        __syncthreads();
    }
    m = red[0]; __syncthreads();

    float sum = 0.f;
    for (int j = t; j < N_NODES; j += 256) {
        float e = (arow[j] == 1) ? expf(lrow[j] - m) : 0.f;
        lrow[j] = e;
        sum += e;
    }
    red[t] = sum; __syncthreads();
    for (int s = 128; s > 0; s >>= 1) {
        if (t < s) red[t] += red[t + s];
        __syncthreads();
    }
    const float inv = 1.f / red[0];
    for (int j = t; j < N_NODES; j += 256) lrow[j] *= inv;
}

// ---------------------------------------------------------------------------
// Kernel D: nbc = attn @ embs; out = concat([embs, nbc], axis=1).
// ---------------------------------------------------------------------------
__global__ __launch_bounds__(128) void output_kernel(
    const float* __restrict__ embs, const float* __restrict__ attn,
    float* __restrict__ out)
{
    const int i = blockIdx.x;
    const int l = threadIdx.x;
    const float* ar = attn + (size_t)i * N_NODES;

    float acc = 0.f;
    #pragma unroll 4
    for (int j = 0; j < N_NODES; ++j)
        acc = fmaf(ar[j], embs[j * E_DIM + l], acc);

    out[(size_t)i * 2 * E_DIM + l]         = embs[i * E_DIM + l];
    out[(size_t)i * 2 * E_DIM + E_DIM + l] = acc;
}

// ---------------------------------------------------------------------------
extern "C" void kernel_launch(void* const* d_in, const int* in_sizes, int n_in,
                              void* d_out, int out_size, void* d_ws, size_t ws_size,
                              hipStream_t stream)
{
    (void)in_sizes; (void)n_in; (void)out_size; (void)ws_size;

    const float* embs = (const float*)d_in[0];
    const int*   adj  = (const int*)  d_in[1];
    const float* pde  = (const float*)d_in[2];
    const float* W    = (const float*)d_in[3];
    const float* pc_W = (const float*)d_in[4];
    const float* pc_b = (const float*)d_in[5];
    const float* a_W  = (const float*)d_in[6];
    const float* a_b  = (const float*)d_in[7];
    float* out = (float*)d_out;

    char* ws = (char*)d_ws;
    float* cb     = (float*)(ws);                        // 1024*128 f32 = 512 KB
    float* wfa1   = (float*)(ws + (512u << 10));         // 1024 f32
    float* Wa2    = (float*)(ws + (512u << 10) + 4096);  // 128 f32
    float* logits = (float*)(ws + (512u << 10) + 8192);  // 1024*1024 f32 = 4 MB

    const float* pc_Wp = pc_W + E_DIM * E_DIM;           // pc_W[E:, :]

    setup_kernel<<<dim3(N_NODES), dim3(E_DIM), 0, stream>>>(
        embs, W, pc_W, pc_b, a_W, cb, wfa1, Wa2);

    logits_kernel<<<dim3(N_NODES / JT, N_NODES), dim3(256), 0, stream>>>(
        pde, pc_Wp, cb, wfa1, Wa2, a_b, logits);

    softmax_kernel<<<dim3(N_NODES), dim3(256), 0, stream>>>(adj, logits);

    output_kernel<<<dim3(N_NODES), dim3(E_DIM), 0, stream>>>(embs, logits, out);
}
